// _BaselineLSTMEncoder_12927851561572
// MI455X (gfx1250) — matrix operand order repaired and checked by reading, not compiled, not run on hardware
//
#include <hip/hip_runtime.h>

#define N_BATCH 1024
#define C_IN    128
#define T_STEPS 256
#define H_DIM   512
#define G4      (4 * H_DIM)   // 2048 gate columns
#define HPITCH  520           // LDS row pitch in halves (+16B pad: kills bank conflicts)

typedef __attribute__((ext_vector_type(16))) __bf16       v16bf;
typedef __attribute__((ext_vector_type(8)))  float        v8f;
typedef __attribute__((ext_vector_type(4)))  unsigned int v4u;

union Frag {
    v16bf v;
    v4u   q[2];
};

static __device__ __forceinline__ unsigned short f2bf(float f) {
    unsigned int u = __float_as_uint(f);
    u += 0x7fffu + ((u >> 16) & 1u);   // round-to-nearest-even
    return (unsigned short)(u >> 16);
}

static __device__ __forceinline__ float fast_rcp(float x) {
#if __has_builtin(__builtin_amdgcn_rcpf)
    return __builtin_amdgcn_rcpf(x);   // v_rcp_f32
#else
    return 1.0f / x;
#endif
}

static __device__ __forceinline__ float sigm(float x) {
    return fast_rcp(1.0f + __expf(-x));          // v_exp_f32 + v_rcp_f32
}

static __device__ __forceinline__ float tanh_fast(float x) {
    const float e = __expf(-2.0f * x);           // tanh = (1-e)/(1+e)
    return (1.0f - e) * fast_rcp(1.0f + e);
}

// ---------------- prep kernel 1: weights -> bf16, bias sum -------------------
__global__ void prep_weights(const float* __restrict__ Whh, const float* __restrict__ Wih,
                             const float* __restrict__ bih, const float* __restrict__ bhh,
                             unsigned short* __restrict__ Whh_b,
                             unsigned short* __restrict__ Wih_b,
                             float* __restrict__ bsum) {
    int idx = blockIdx.x * blockDim.x + threadIdx.x;
    const int nWhh = G4 * H_DIM;           // 1048576
    const int nWih = G4 * C_IN;            // 262144
    if (idx < nWhh) {
        Whh_b[idx] = f2bf(Whh[idx]);
    } else if (idx < nWhh + nWih) {
        int i = idx - nWhh;
        Wih_b[i] = f2bf(Wih[i]);
    } else if (idx < nWhh + nWih + G4) {
        int g = idx - nWhh - nWih;
        bsum[g] = bih[g] + bhh[g];
    }
}

// ---------------- prep kernel 2: x [N,C,T] f32 -> xT [T,N,C] bf16 ------------
__global__ void prep_xT(const float* __restrict__ x, unsigned short* __restrict__ xT) {
    int idx = blockIdx.x * blockDim.x + threadIdx.x;   // linear over [T][N][C]
    if (idx >= T_STEPS * N_BATCH * C_IN) return;
    int c = idx & (C_IN - 1);
    int n = (idx >> 7) & (N_BATCH - 1);
    int t = idx >> 17;
    xT[idx] = f2bf(x[(n * C_IN + c) * T_STEPS + t]);
}

// ---------------- fused persistent LSTM kernel -------------------------------
// grid: 64 WGs (16 batch rows each), 512 threads = 16 wave32s.
// Wave w owns hidden columns [32w, 32w+32): 2 column tiles x 4 gates
// = 8 live 16x16 f32 WMMA accumulators (64 VGPRs).
// Weight fragments stream from L2 every step. A laundered SGPR zero-offset
// (not a laundered pointer!) keeps LICM from hoisting the t-invariant weight
// loads while preserving pointer provenance -> global_load_b128, not flat.
__global__ __launch_bounds__(512) void lstm_fused(
    const unsigned short* __restrict__ Whh_b,   // [2048][512] bf16
    const unsigned short* __restrict__ Wih_b,   // [2048][128] bf16
    const float*          __restrict__ bsum,    // [2048]
    const unsigned short* __restrict__ xT,      // [256][1024][128] bf16
    float*                __restrict__ out)     // [1024][512]
{
    __shared__ unsigned short hbuf[2][16][HPITCH];   // double-buffered h tile (bf16)

    const int tid   = threadIdx.x;
    const int wave  = tid >> 5;          // 0..15
    const int lane  = tid & 31;
    const int lrow  = lane & 15;         // M row (A/C frags) / N column (B/C frags)
    const int lhi   = lane >> 4;         // K-half selector per ISA VGPR layouts
    const int nbase = blockIdx.x * 16;
    const int wbase = wave * 32;         // first hidden column owned by this wave

    // zero initial h state (both buffers)
    for (int i = tid; i < 2 * 16 * HPITCH; i += 512)
        ((unsigned short*)hbuf)[i] = 0;

    // Per-lane weight base pointers; per-accumulator deltas are immediates.
    // u = gate*2 + tl:  gcol = gate*512 + wbase + tl*16 + lrow
    const unsigned short* whBase = Whh_b + (size_t)(wbase + lrow) * H_DIM + lhi * 8;
    const unsigned short* wiBase = Wih_b + (size_t)(wbase + lrow) * C_IN  + lhi * 8;

    // Bias per accumulator (C-frag: all 8 elems of a lane share N = lane&15)
    float bs[8];
#pragma unroll
    for (int u = 0; u < 8; ++u) {
        const int gate = u >> 1, tl = u & 1;
        bs[u] = bsum[gate * H_DIM + wbase + tl * 16 + lrow];
    }

    v8f cst[2];   // cell state fragments (fp32, resident in registers all T steps)
#pragma unroll
    for (int tl = 0; tl < 2; ++tl) {
        v8f z = {0.f, 0.f, 0.f, 0.f, 0.f, 0.f, 0.f, 0.f};
        cst[tl] = z;
    }

    __syncthreads();

#pragma unroll 1
    for (int t = 0; t < T_STEPS; ++t) {
        const int cur = t & 1, nxt = cur ^ 1;

        // Laundered zero offset in an SGPR: redefined every iteration, so the
        // weight loads below are not loop-invariant (no hoist -> no spill),
        // but the pointer provenance (global addrspace) is preserved.
        size_t zoff = 0;
        asm volatile("" : "+s"(zoff));
        const unsigned short* wiB = wiBase + zoff;
        const unsigned short* whB = whBase + zoff;

        v8f acc[8];
#pragma unroll
        for (int u = 0; u < 8; ++u) {
            const float b = bs[u];
            v8f a = {b, b, b, b, b, b, b, b};
            acc[u] = a;
        }

        // ---- input projection: K = C_IN = 128 -> 4 chunks of 32 ----
        const unsigned short* xrow =
            xT + ((size_t)t * N_BATCH + nbase + lrow) * C_IN;
        for (int kc = 0; kc < 4; ++kc) {
            Frag a;
            const int kb = kc * 32 + lhi * 8;
            a.q[0] = *(const v4u*)(xrow + kb);        // K kb..kb+7
            a.q[1] = *(const v4u*)(xrow + kb + 16);   // K kb+16..kb+23
            const unsigned short* pk = wiB + kc * 32;
#pragma unroll
            for (int u = 0; u < 8; ++u) {
                const int off = ((u >> 1) * H_DIM + (u & 1) * 16) * C_IN; // imm
                Frag b;
                b.q[0] = *(const v4u*)(pk + off);
                b.q[1] = *(const v4u*)(pk + off + 16);
                acc[u] = __builtin_amdgcn_wmma_f32_16x16x32_bf16(
                    false, a.v, false, b.v, (short)0, acc[u], false, false);
            }
        }

        // ---- recurrent projection: K = H = 512 -> 16 chunks of 32 ----
        for (int kc = 0; kc < 16; ++kc) {
            Frag a;
            const int kb = kc * 32 + lhi * 8;
            a.q[0] = *(const v4u*)(&hbuf[cur][lrow][kb]);        // ds_load_b128
            a.q[1] = *(const v4u*)(&hbuf[cur][lrow][kb + 16]);
            const unsigned short* pk = whB + kc * 32;
#pragma unroll
            for (int u = 0; u < 8; ++u) {
                const int off = ((u >> 1) * H_DIM + (u & 1) * 16) * H_DIM; // imm
                Frag b;
                b.q[0] = *(const v4u*)(pk + off);      // global_load_b128, L2-resident
                b.q[1] = *(const v4u*)(pk + off + 16);
                acc[u] = __builtin_amdgcn_wmma_f32_16x16x32_bf16(
                    false, a.v, false, b.v, (short)0, acc[u], false, false);
            }
        }

        // ---- elementwise LSTM cell update on C/D fragment layout ----
        // C layout: vgpr e, lanes 0-15 -> (M=e, N=lane); lanes 16-31 -> (M=8+e)
#pragma unroll
        for (int tl = 0; tl < 2; ++tl) {
            v8f gi = acc[tl], gf = acc[2 + tl], gg = acc[4 + tl], go = acc[6 + tl];
            const int col = wbase + tl * 16 + lrow;
#pragma unroll
            for (int e = 0; e < 8; ++e) {
                const float cn = sigm(gf[e]) * cst[tl][e] + sigm(gi[e]) * tanh_fast(gg[e]);
                const float hn = sigm(go[e]) * tanh_fast(cn);
                cst[tl][e] = cn;
                const int row = e + lhi * 8;
                if (t == T_STEPS - 1) {
                    out[(size_t)(nbase + row) * H_DIM + col] = hn;
                } else {
                    hbuf[nxt][row][col] = f2bf(hn);
                }
            }
        }
        __syncthreads();   // h(t+1) tile visible to all 16 waves for next step
    }
}

// ---------------- host launch ------------------------------------------------
extern "C" void kernel_launch(void* const* d_in, const int* in_sizes, int n_in,
                              void* d_out, int out_size, void* d_ws, size_t ws_size,
                              hipStream_t stream) {
    const float* x   = (const float*)d_in[0];   // [N, C, T]
    const float* Wih = (const float*)d_in[1];   // [4H, C]
    const float* Whh = (const float*)d_in[2];   // [4H, H]
    const float* bih = (const float*)d_in[3];   // [4H]
    const float* bhh = (const float*)d_in[4];   // [4H]
    float* out = (float*)d_out;                 // [N, H]

    char* ws = (char*)d_ws;
    unsigned short* Whh_b = (unsigned short*)(ws);                 // 2,097,152 B
    unsigned short* Wih_b = (unsigned short*)(ws + 2097152);       //   524,288 B
    float*          bsum  = (float*)(ws + 2621440);                //     8,192 B
    unsigned short* xTb   = (unsigned short*)(ws + 2629632);       // 67,108,864 B

    const int nprep = G4 * H_DIM + G4 * C_IN + G4;
    prep_weights<<<(nprep + 255) / 256, 256, 0, stream>>>(Whh, Wih, bih, bhh,
                                                          Whh_b, Wih_b, bsum);
    const int nxT = T_STEPS * N_BATCH * C_IN;
    prep_xT<<<(nxT + 255) / 256, 256, 0, stream>>>(x, xTb);

    lstm_fused<<<N_BATCH / 16, 512, 0, stream>>>(Whh_b, Wih_b, bsum, xTb, out);
}
